// scale_block_80401787781844
// MI455X (gfx1250) — compile-verified, matrix operand order checked by reading, not run on hardware
//
#include <hip/hip_runtime.h>
#include <hip/hip_bf16.h>
#include <stdint.h>

// Fused LayerNorm + GEMM ("seg_merge") for MI455X / gfx1250.
//   A[M=256*513, K=1024] built on the fly from x (layernormed, bf16),
//   B = lin_w [1024, 512] prepacked to bf16 WMMA fragments in d_ws (1 MB),
//   C = A @ B + lin_b, f32 accumulation via v_wmma_f32_16x16x32_bf16.

typedef __attribute__((ext_vector_type(16))) __bf16 v16bf;
typedef __attribute__((ext_vector_type(8)))  float  v8f;

#define S_SLICES   256
#define L_SEQ      1025
#define D_MODEL    512
#define K_DIM      1024
#define M_ROWS     513
#define MTILE      32
#define MTILES     17          // ceil(513/32)
#define EPS_LN     1e-5f
#define LDS_ROW_U16 1032       // 1024 bf16 + 8 pad -> 2064B row stride (4-bank rotation)

__device__ __forceinline__ unsigned short f2bf(float f) {
    union { float f; uint32_t u; } c; c.f = f;
    uint32_t u = c.u;
    uint32_t r = u + 0x7FFFu + ((u >> 16) & 1u);   // round-to-nearest-even
    return (unsigned short)(r >> 16);
}

// ---------------------------------------------------------------------------
// Prepack lin_w [K=1024][N=512] f32 -> bf16 B-fragments.
// Fragment (nt, kt): 32x16 (KxN) tile; lane l holds col n = nt*16 + (l&15),
// K values kt*32 + {8*(l>>4)+0..7, 16+8*(l>>4)+0..7}  (ISA 16-bit B layout).
// Stored contiguously: bpack[((nt*32+kt)*32 + lane)*8 ..+8] (8 uints = 16 bf16).
// ---------------------------------------------------------------------------
__global__ __launch_bounds__(256) void prepack_b_kernel(
    const float* __restrict__ lin_w, uint32_t* __restrict__ bpack)
{
    const int lane = threadIdx.x & 31;
    const int frag = blockIdx.x * 8 + (threadIdx.x >> 5);   // 0..1023
    const int nt = frag >> 5, kt = frag & 31;
    const int half = lane >> 4;
    const int n = nt * 16 + (lane & 15);

    uint32_t vals[8];
#pragma unroll
    for (int e2 = 0; e2 < 8; ++e2) {
        const int e0 = 2 * e2;                               // even element
        const int k0 = kt * 32 + 16 * (e0 >> 3) + 8 * half + (e0 & 7);
        const unsigned short lo = f2bf(lin_w[(size_t)k0 * D_MODEL + n]);
        const unsigned short hi = f2bf(lin_w[(size_t)(k0 + 1) * D_MODEL + n]);
        vals[e2] = (uint32_t)lo | ((uint32_t)hi << 16);
    }
    uint32_t* dst = bpack + ((size_t)frag * 32 + lane) * 8;
    *(uint4*)(dst)     = make_uint4(vals[0], vals[1], vals[2], vals[3]);
    *(uint4*)(dst + 4) = make_uint4(vals[4], vals[5], vals[6], vals[7]);
}

// ---------------------------------------------------------------------------
// Main fused kernel. Grid: (17 m-tiles, 256 slices). Block: 256 (8 wave32).
// ---------------------------------------------------------------------------
__global__ __launch_bounds__(256) void segmerge_ln_gemm_kernel(
    const float*    __restrict__ x,
    const float*    __restrict__ norm_w,
    const float*    __restrict__ norm_b,
    const float*    __restrict__ lin_b,
    const uint32_t* __restrict__ bpack,
    float*          __restrict__ out)
{
    __shared__ unsigned short ldsA[MTILE * LDS_ROW_U16];     // 66048 B

    const int slice = blockIdx.y;
    const int mtile = blockIdx.x;
    const int tid   = threadIdx.x;
    const int wave  = tid >> 5;
    const int lane  = tid & 31;
    const int half  = lane >> 4;
    const int l16   = lane & 15;

    const float* xs = x + (size_t)slice * (L_SEQ * D_MODEL);

    // ---- Phase 1: LayerNorm 32 rows -> bf16 in LDS (one wave per row) ----
    for (int r = wave; r < MTILE; r += 8) {
        const int R = mtile * MTILE + r;
        unsigned short* dstrow = ldsA + r * LDS_ROW_U16;
        if (R < M_ROWS) {
            // row R>=1: 1024 contiguous floats (two adjacent L-rows);
            // row 0: 512 cls floats duplicated along features.
            const float* rowp = (R == 0) ? xs : (xs + (size_t)(2 * R - 1) * D_MODEL);
            float4 v[8];
            float s = 0.f, ss = 0.f;
#pragma unroll
            for (int it = 0; it < 8; ++it) {
                const int f  = it * 128 + lane * 4;
                const int f2 = (R == 0) ? (f & 511) : f;
                const float4 t = *(const float4*)(rowp + f2);
                v[it] = t;
                s  += t.x + t.y + t.z + t.w;
                ss += t.x * t.x + t.y * t.y + t.z * t.z + t.w * t.w;
            }
#pragma unroll
            for (int m = 16; m >= 1; m >>= 1) {
                s  += __shfl_xor(s,  m, 32);
                ss += __shfl_xor(ss, m, 32);
            }
            const float mean = s * (1.0f / K_DIM);
            const float var  = ss * (1.0f / K_DIM) - mean * mean;
            const float rstd = rsqrtf(var + EPS_LN);
#pragma unroll
            for (int it = 0; it < 8; ++it) {
                const int f = it * 128 + lane * 4;
                const float4 w = *(const float4*)(norm_w + f);
                const float4 b = *(const float4*)(norm_b + f);
                const float4 t = v[it];
                const float o0 = (t.x - mean) * rstd * w.x + b.x;
                const float o1 = (t.y - mean) * rstd * w.y + b.y;
                const float o2 = (t.z - mean) * rstd * w.z + b.z;
                const float o3 = (t.w - mean) * rstd * w.w + b.w;
                uint2 p;
                p.x = (uint32_t)f2bf(o0) | ((uint32_t)f2bf(o1) << 16);
                p.y = (uint32_t)f2bf(o2) | ((uint32_t)f2bf(o3) << 16);
                *(uint2*)(dstrow + f) = p;
            }
        } else {
            // pad rows (tail tile): zero so WMMA stays defined; stores are guarded.
#pragma unroll
            for (int it = 0; it < 8; ++it) {
                const int f = it * 128 + lane * 4;
                *(uint2*)(dstrow + f) = make_uint2(0u, 0u);
            }
        }
    }
    __syncthreads();

    // ---- Phase 2: WMMA GEMM. Wave owns cols [wave*64, wave*64+64) x 32 rows.
    v8f acc[8];
#pragma unroll
    for (int i = 0; i < 8; ++i) acc[i] = (v8f)(0.0f);

    union Frag { uint4 q[2]; v16bf v; };

    for (int kt = 0; kt < 32; ++kt) {
        // A fragments (16x32 bf16) for mt=0,1 from LDS.
        // lane: row = l16 (+16 for mt=1), K base = kt*32 + 8*half; halves at +0,+16.
        Frag afrag[2];
#pragma unroll
        for (int mt = 0; mt < 2; ++mt) {
            const unsigned short* ap =
                ldsA + (mt * 16 + l16) * LDS_ROW_U16 + kt * 32 + 8 * half;
            afrag[mt].q[0] = *(const uint4*)(ap);
            afrag[mt].q[1] = *(const uint4*)(ap + 16);
        }
#pragma unroll
        for (int j = 0; j < 4; ++j) {
            const int nt = wave * 4 + j;
            const uint32_t* bpp = bpack + ((size_t)(nt * 32 + kt) * 32 + lane) * 8;
            Frag bfrag;
            bfrag.q[0] = *(const uint4*)(bpp);
            bfrag.q[1] = *(const uint4*)(bpp + 4);
#pragma unroll
            for (int mt = 0; mt < 2; ++mt) {
                acc[j * 2 + mt] = __builtin_amdgcn_wmma_f32_16x16x32_bf16(
                    false, afrag[mt].v, false, bfrag.v,
                    (short)0, acc[j * 2 + mt], false, false);
            }
        }
    }

    // ---- Phase 3: bias + store. C/D layout: vgpr i, lane -> row i+8*half, col l16.
#pragma unroll
    for (int j = 0; j < 4; ++j) {
        const int n = wave * 64 + j * 16 + l16;
        const float bias = lin_b[n];
#pragma unroll
        for (int mt = 0; mt < 2; ++mt) {
#pragma unroll
            for (int i = 0; i < 8; ++i) {
                const int R = mtile * MTILE + mt * 16 + i + 8 * half;
                if (R < M_ROWS) {
                    out[((size_t)slice * M_ROWS + R) * D_MODEL + n] =
                        acc[j * 2 + mt][i] + bias;
                }
            }
        }
    }
}

// ---------------------------------------------------------------------------
extern "C" void kernel_launch(void* const* d_in, const int* in_sizes, int n_in,
                              void* d_out, int out_size, void* d_ws, size_t ws_size,
                              hipStream_t stream) {
    const float* x      = (const float*)d_in[0];   // [8,32,1025,512]
    const float* norm_w = (const float*)d_in[1];   // [1024]
    const float* norm_b = (const float*)d_in[2];   // [1024]
    const float* lin_w  = (const float*)d_in[3];   // [1024,512]
    const float* lin_b  = (const float*)d_in[4];   // [512]
    float* out = (float*)d_out;                    // [8,32,513,512]
    uint32_t* bpack = (uint32_t*)d_ws;             // needs 1 MB scratch

    // Prepack B into bf16 WMMA fragments (1024 fragments, 8 per 256-thread block).
    prepack_b_kernel<<<128, 256, 0, stream>>>(lin_w, bpack);

    dim3 grid(MTILES, S_SLICES);                   // (17, 256)
    segmerge_ln_gemm_kernel<<<grid, 256, 0, stream>>>(
        x, norm_w, norm_b, lin_b, bpack, out);
}
